// Model_90537910599720
// MI455X (gfx1250) — matrix operand order repaired and checked by reading, not compiled, not run on hardware
//
#include <hip/hip_runtime.h>
#include <cstdint>
#include <cstddef>

// ---------------------------------------------------------------------------
// S-Mamba forecaster forward for MI455X (gfx1250, wave32, WMMA).
//
// GEMMs run on v_wmma_f32_16x16x32_bf16 with split-bf16 operands
// (x = hi + lo; product = hi*hi + hi*lo + lo*hi -> ~fp32 accuracy).
// Tiles are staged global->LDS with GLOBAL_LOAD_ASYNC_TO_LDS_B128 (ASYNCcnt),
// double-buffered so tile k+1's DMA overlaps tile k's WMMAs.
// Selective scan parallelized over (batch, d) = 8192 lanes, 16 states each.
// ---------------------------------------------------------------------------

typedef __attribute__((ext_vector_type(16))) __bf16 v16bf;
typedef __attribute__((ext_vector_type(8)))  float  v8f;
typedef int v4i __attribute__((ext_vector_type(4)));

#define BATCH   16
#define SEQ     512
#define NVARS   256
#define DM      512
#define DSTATE  16
#define DTRANK  32
#define DFF     2048
#define PRED    96
#define TOKENS  (BATCH*NVARS)   // 4096 tokens, each a 512-dim vector

#ifndef __has_builtin
#define __has_builtin(x) 0
#endif
#if __has_builtin(__builtin_amdgcn_global_load_async_to_lds_b128) && \
    __has_builtin(__builtin_amdgcn_s_wait_asynccnt)
#define HAVE_ASYNC_LDS 1
#else
#define HAVE_ASYNC_LDS 0
#endif

#if HAVE_ASYNC_LDS
typedef __attribute__((address_space(1))) v4i as1_v4i;   // global
typedef __attribute__((address_space(3))) v4i as3_v4i;   // LDS
__device__ __forceinline__ void async_copy16(const float* g, float* l) {
    // one B128 (16 bytes) per lane, global -> LDS, tracked by ASYNCcnt.
    // AS-qualified pointers convert implicitly to generic if needed.
    __builtin_amdgcn_global_load_async_to_lds_b128((as1_v4i*)g, (as3_v4i*)l, 0, 0);
}
#endif

__device__ __forceinline__ void split2(float x, __bf16& hi, __bf16& lo) {
    float fh = __uint_as_float(__float_as_uint(x) & 0xffff0000u); // truncate
    hi = (__bf16)fh;          // exact (fh is a bf16 value)
    lo = (__bf16)(x - fh);    // residual, RNE
}
__device__ __forceinline__ float silu_f(float x) { return x / (1.f + __expf(-x)); }

// ---------------------------------------------------------------------------
// WMMA GEMM: C[M,N] = A[M,K] @ B[K,N] (+bias)(+act)(+=C)
// Block tile 64x64, BK=32, 128 threads = 4 waves, each wave 2x2 WMMA tiles.
// Requires: M % 64 == 0, K % 32 == 0 (true for all calls here). N guarded.
// ---------------------------------------------------------------------------
#define BM 64
#define BN 64
#define BK 32
#define ASTR 36   // A row stride (floats): 144B, 16B-aligned, conflict-spread
#define BSTR 68   // B row stride (floats): 272B, 16B-aligned

template<int HAS_BIAS, int ACT, int ACCUM>   // ACT: 0 none, 1 relu, 2 softplus
__global__ __launch_bounds__(128)
void gemm_wmma(const float* __restrict__ A, int lda,
               const float* __restrict__ B, int ldb,
               float* __restrict__ C, int ldc,
               const float* __restrict__ bias,
               int M, int N, int K)
{
    __shared__ float As[2][BM][ASTR];   // fp32 tiles, double buffered
    __shared__ float Bs[2][BK][BSTR];

    const int tid  = threadIdx.x;
    const int lane = tid & 31;
    const int wave = tid >> 5;
    const int hlf  = lane >> 4;     // 0: lanes 0-15, 1: lanes 16-31
    const int l16  = lane & 15;

    const int row0 = blockIdx.y * BM;
    const int col0 = blockIdx.x * BN;
    const int wm   = (wave >> 1) * 32;   // wave quadrant in block tile
    const int wn   = (wave & 1) * 32;
    const bool interior = (col0 + BN <= N);   // block-uniform

    // staging coordinates (128 threads, 16B per thread per pass)
    const int ar = tid >> 3;          // A: rows 0..15 (+16 per pass), 4 passes
    const int ac = (tid & 7) * 4;
    const int bk = tid >> 4;          // B: k-rows 0..7 (+8 per pass), 4 passes
    const int bn = (tid & 15) * 4;

    auto stage = [&](int buf, int k0) {
#if HAVE_ASYNC_LDS
        if (interior) {
            #pragma unroll
            for (int r = 0; r < 4; ++r) {
                const float* g = A + (size_t)(row0 + r*16 + ar) * lda + k0 + ac;
                async_copy16(g, &As[buf][r*16 + ar][ac]);
            }
            #pragma unroll
            for (int r = 0; r < 4; ++r) {
                const float* g = B + (size_t)(k0 + r*8 + bk) * ldb + col0 + bn;
                async_copy16(g, &Bs[buf][r*8 + bk][bn]);
            }
            return;
        }
#endif
        #pragma unroll
        for (int r = 0; r < 4; ++r) {
            const float* g = A + (size_t)(row0 + r*16 + ar) * lda + k0 + ac;
            float4 v4 = *(const float4*)g;
            float* d = &As[buf][r*16 + ar][ac];
            d[0]=v4.x; d[1]=v4.y; d[2]=v4.z; d[3]=v4.w;
        }
        #pragma unroll
        for (int r = 0; r < 4; ++r) {
            int gn = col0 + bn;
            const float* g = B + (size_t)(k0 + r*8 + bk) * ldb + gn;
            float x0, x1, x2, x3;
            if (gn + 3 < N) {
                float4 v4 = *(const float4*)g;
                x0=v4.x; x1=v4.y; x2=v4.z; x3=v4.w;
            } else {
                x0 = (gn+0 < N) ? g[0] : 0.f;
                x1 = (gn+1 < N) ? g[1] : 0.f;
                x2 = (gn+2 < N) ? g[2] : 0.f;
                x3 = (gn+3 < N) ? g[3] : 0.f;
            }
            float* d = &Bs[buf][r*8 + bk][bn];
            d[0]=x0; d[1]=x1; d[2]=x2; d[3]=x3;
        }
    };

    const v8f vzero = {0.f,0.f,0.f,0.f,0.f,0.f,0.f,0.f};
    v8f acc[2][2];
    acc[0][0] = vzero; acc[0][1] = vzero; acc[1][0] = vzero; acc[1][1] = vzero;

    stage(0, 0);
    const int nk = K / BK;
    for (int it = 0; it < nk; ++it) {
        const int cur = it & 1;
        const bool has_next = (it + 1 < nk);
        if (has_next) stage(1 - cur, (it + 1) * BK);   // prefetch next tile
#if HAVE_ASYNC_LDS
        if (interior) {
            // allow the 8 just-issued (next-tile) copies to remain in flight;
            // current tile's copies are guaranteed complete (in-order returns)
            if (has_next) __builtin_amdgcn_s_wait_asynccnt(8);
            else          __builtin_amdgcn_s_wait_asynccnt(0);
        }
#endif
        __syncthreads();

        // ---- build fragments per ISA VGPR layouts, split fp32 -> bf16 hi/lo
        // A 16x32: lanes 0-15 -> M=lane, K pairs {0..7,16..23}; lanes 16-31 +8
        v16bf ahf[2], alf[2], bhf[2], blf[2];
        #pragma unroll
        for (int tm = 0; tm < 2; ++tm) {
            int m = wm + tm * 16 + l16;
            #pragma unroll
            for (int q = 0; q < 8; ++q) {
                int kk = (q & 3) * 2 + (q >> 2) * 16 + (hlf ? 8 : 0);
                __bf16 h, l;
                split2(As[cur][m][kk],   h, l); ahf[tm][2*q+0] = h; alf[tm][2*q+0] = l;
                split2(As[cur][m][kk+1], h, l); ahf[tm][2*q+1] = h; alf[tm][2*q+1] = l;
            }
        }
        // B 32x16: lanes 0-15 -> N=lane, K {0..7,16..23}; lanes 16-31 +8
        #pragma unroll
        for (int tn = 0; tn < 2; ++tn) {
            int n = wn + tn * 16 + l16;
            #pragma unroll
            for (int q = 0; q < 8; ++q) {
                int kk = (q & 3) * 2 + (q >> 2) * 16 + (hlf ? 8 : 0);
                __bf16 h, l;
                split2(Bs[cur][kk][n],   h, l); bhf[tn][2*q+0] = h; blf[tn][2*q+0] = l;
                split2(Bs[cur][kk+1][n], h, l); bhf[tn][2*q+1] = h; blf[tn][2*q+1] = l;
            }
        }
        // ---- 3 WMMAs per tile: hi*hi + hi*lo + lo*hi ~= fp32 product ----
        #pragma unroll
        for (int tm = 0; tm < 2; ++tm)
            #pragma unroll
            for (int tn = 0; tn < 2; ++tn) {
                acc[tm][tn] = __builtin_amdgcn_wmma_f32_16x16x32_bf16(
                    false, ahf[tm], false, bhf[tn], (short)0, acc[tm][tn], false, false);
                acc[tm][tn] = __builtin_amdgcn_wmma_f32_16x16x32_bf16(
                    false, ahf[tm], false, blf[tn], (short)0, acc[tm][tn], false, false);
                acc[tm][tn] = __builtin_amdgcn_wmma_f32_16x16x32_bf16(
                    false, alf[tm], false, bhf[tn], (short)0, acc[tm][tn], false, false);
            }
        __syncthreads();   // all waves done reading buf[cur] before it is re-staged
    }

    // ---- epilogue: C/D layout VGPR r -> M=r (lanes 0-15) / M=r+8 (lanes 16-31)
    #pragma unroll
    for (int tm = 0; tm < 2; ++tm)
        #pragma unroll
        for (int tn = 0; tn < 2; ++tn) {
            int gm0 = row0 + wm + tm * 16 + (hlf ? 8 : 0);
            int gn  = col0 + wn + tn * 16 + l16;
            if (gn < N) {
                float badd = HAS_BIAS ? bias[gn] : 0.f;
                #pragma unroll
                for (int r = 0; r < 8; ++r) {
                    float val = acc[tm][tn][r] + badd;
                    if (ACT == 1) val = fmaxf(val, 0.f);
                    if (ACT == 2) val = (val > 20.f) ? val : log1pf(__expf(val));
                    float* p = C + (size_t)(gm0 + r) * ldc + gn;
                    if (ACCUM) val += *p;
                    *p = val;
                }
            }
        }
}

// ---------------------------------------------------------------------------
// Per-(batch,var) masked normalization stats + transposed normalized input.
// xnT[(b*256+v)][j] = (x[b,j,v]-mean)/std * valid
// ---------------------------------------------------------------------------
__global__ __launch_bounds__(128)
void stats_kernel(const float* __restrict__ x, float* __restrict__ means,
                  float* __restrict__ stdev, float* __restrict__ xnT)
{
    __shared__ float r1[128], r2[128];
    __shared__ float st[2];
    int row = blockIdx.x;            // b*256 + v
    int b = row >> 8, v = row & 255;
    int tid = threadIdx.x;
    const float* xp = x + (size_t)b * SEQ * NVARS + v;
    float vals[4], vld[4];
    float s = 0.f, c = 0.f;
    #pragma unroll
    for (int i = 0; i < 4; ++i) {
        int j = tid + i * 128;
        float val = xp[(size_t)j * NVARS];
        float w = (v == 0) ? 1.f : ((val != 0.f) ? 1.f : 0.f);
        vals[i] = val; vld[i] = w; s += val * w; c += w;
    }
    r1[tid] = s; r2[tid] = c; __syncthreads();
    for (int off = 64; off; off >>= 1) {
        if (tid < off) { r1[tid] += r1[tid+off]; r2[tid] += r2[tid+off]; }
        __syncthreads();
    }
    if (tid == 0) { st[1] = r2[0] + 1e-8f; st[0] = r1[0] / st[1]; }
    __syncthreads();
    float mean = st[0], cnt = st[1];
    float sq = 0.f;
    #pragma unroll
    for (int i = 0; i < 4; ++i) { float d = vals[i] - mean; sq += d * d * vld[i]; }
    __syncthreads();
    r1[tid] = sq; __syncthreads();
    for (int off = 64; off; off >>= 1) {
        if (tid < off) r1[tid] += r1[tid+off];
        __syncthreads();
    }
    if (tid == 0) st[0] = sqrtf(r1[0] / cnt + 1e-8f);
    __syncthreads();
    float sd = st[0], inv = 1.f / sd;
    #pragma unroll
    for (int i = 0; i < 4; ++i) {
        int j = tid + i * 128;
        xnT[(size_t)row * SEQ + j] = (vals[i] - mean) * inv * vld[i];
    }
    if (tid == 0) { means[row] = mean; stdev[row] = sd; }
}

// mean over sequence of the 6 time features, per batch
__global__ void xm_mean_kernel(const float* __restrict__ xm, float* __restrict__ out)
{
    int b = blockIdx.x, c = threadIdx.x;
    if (c < 6) {
        float s = 0.f;
        for (int j = 0; j < SEQ; ++j) s += xm[(size_t)b * SEQ * 6 + j * 6 + c];
        out[b * 6 + c] = s * (1.f / (float)SEQ);
    }
}

// enc[row][n] += (mean_j x_mark) @ Wt + bt
__global__ __launch_bounds__(256)
void add_time_kernel(float* __restrict__ enc, const float* __restrict__ xmm,
                     const float* __restrict__ Wt, const float* __restrict__ bt)
{
    int idx = blockIdx.x * 256 + threadIdx.x;
    int n = idx & (DM - 1);
    int b = idx >> (9 + 8);                   // row = idx>>9 ; b = row>>8
    float s = bt[n];
    #pragma unroll
    for (int c = 0; c < 6; ++c) s += xmm[b * 6 + c] * Wt[c * DM + n];
    enc[idx] += s;
}

// depthwise width-2 conv + SiLU; dir=0: neighbor t-1 (fwd), dir=1: t+1 (rev)
__global__ __launch_bounds__(256)
void conv_silu_kernel(const float* __restrict__ xz, const float* __restrict__ cw,
                      const float* __restrict__ cb, float* __restrict__ uc, int dir)
{
    int idx = blockIdx.x * 256 + threadIdx.x;     // TOKENS*DM
    int d = idx & (DM - 1);
    int row = idx >> 9;
    int t = row & (NVARS - 1);
    float u  = xz[(size_t)row * (2 * DM) + d];
    float un = 0.f;
    if (dir == 0) { if (t > 0)         un = xz[(size_t)(row - 1) * (2 * DM) + d]; }
    else          { if (t < NVARS - 1) un = xz[(size_t)(row + 1) * (2 * DM) + d]; }
    float val = un * cw[2 * d] + u * cw[2 * d + 1] + cb[d];
    uc[idx] = silu_f(val);
}

// Selective scan: one thread per (b,d), 16 SSM states in registers, serial over
// L=256 tokens (ascending for fwd, descending for rev). Fuses skip (u*D) and
// SiLU(z) gate, writing y at original token positions (flip-back implicit).
__global__ __launch_bounds__(256)
void scan_kernel(const float* __restrict__ delta, const float* __restrict__ dbc,
                 const float* __restrict__ uc, const float* __restrict__ xz,
                 const float* __restrict__ Alog, const float* __restrict__ Dp,
                 float* __restrict__ y, int dir)
{
    int b = blockIdx.x >> 1;
    int d = (blockIdx.x & 1) * 256 + threadIdx.x;
    float A[DSTATE], h[DSTATE];
    #pragma unroll
    for (int n = 0; n < DSTATE; ++n) { A[n] = -__expf(Alog[d * DSTATE + n]); h[n] = 0.f; }
    float Dd = Dp[d];
    int rowbase = b * NVARS;
    for (int s = 0; s < NVARS; ++s) {
        int t = dir ? (NVARS - 1 - s) : s;
        int row = rowbase + t;
        float dl = delta[(size_t)row * DM + d];
        float u  = uc[(size_t)row * DM + d];
        float du = dl * u;
        const float* bc = dbc + (size_t)row * (DTRANK + 2 * DSTATE);
        float ys = 0.f;
        #pragma unroll
        for (int n = 0; n < DSTATE; ++n) {
            float dA = __expf(dl * A[n]);
            h[n] = dA * h[n] + du * bc[DTRANK + n];          // B_t
            ys += h[n] * bc[DTRANK + DSTATE + n];            // C_t
        }
        float z = xz[(size_t)row * (2 * DM) + DM + d];
        y[(size_t)row * DM + d] = (ys + u * Dd) * silu_f(z);
    }
}

__global__ __launch_bounds__(256)
void layernorm_kernel(const float* __restrict__ x, const float* __restrict__ g,
                      const float* __restrict__ bta, float* __restrict__ out)
{
    __shared__ float red[256];
    __shared__ float st[2];
    int row = blockIdx.x, tid = threadIdx.x;
    const float* xp = x + (size_t)row * DM;
    float a = xp[tid], b = xp[tid + 256];
    red[tid] = a + b; __syncthreads();
    for (int off = 128; off; off >>= 1) { if (tid < off) red[tid] += red[tid+off]; __syncthreads(); }
    if (tid == 0) st[0] = red[0] * (1.f / (float)DM);
    __syncthreads();
    float m = st[0];
    float da = a - m, db = b - m;
    __syncthreads();
    red[tid] = da * da + db * db; __syncthreads();
    for (int off = 128; off; off >>= 1) { if (tid < off) red[tid] += red[tid+off]; __syncthreads(); }
    if (tid == 0) st[1] = rsqrtf(red[0] * (1.f / (float)DM) + 1e-5f);
    __syncthreads();
    float rs = st[1];
    out[(size_t)row * DM + tid]       = da * rs * g[tid]       + bta[tid];
    out[(size_t)row * DM + tid + 256] = db * rs * g[tid + 256] + bta[tid + 256];
}

// out[b][p][v] = dec[(b*256+v)][p] * stdev + mean
__global__ __launch_bounds__(256)
void out_kernel(const float* __restrict__ dec, const float* __restrict__ means,
                const float* __restrict__ stdev, float* __restrict__ out)
{
    int idx = blockIdx.x * 256 + threadIdx.x;
    if (idx >= BATCH * PRED * NVARS) return;
    int v = idx & (NVARS - 1);
    int p = (idx >> 8) % PRED;
    int b = idx / (PRED * NVARS);
    int r = b * NVARS + v;
    out[idx] = dec[(size_t)r * PRED + p] * stdev[r] + means[r];
}

// ---------------------------------------------------------------------------
static void gemm(int hasb, int act, int accum,
                 const float* A, int lda, const float* B, int ldb,
                 float* C, int ldc, const float* bias,
                 int M, int N, int K, hipStream_t s)
{
    dim3 grid((N + BN - 1) / BN, M / BM);
    dim3 blk(128);
    if (accum) {
        if (hasb) gemm_wmma<1,0,1><<<grid,blk,0,s>>>(A,lda,B,ldb,C,ldc,bias,M,N,K);
        else      gemm_wmma<0,0,1><<<grid,blk,0,s>>>(A,lda,B,ldb,C,ldc,bias,M,N,K);
    } else if (act == 1) {
        gemm_wmma<1,1,0><<<grid,blk,0,s>>>(A,lda,B,ldb,C,ldc,bias,M,N,K);
    } else if (act == 2) {
        gemm_wmma<1,2,0><<<grid,blk,0,s>>>(A,lda,B,ldb,C,ldc,bias,M,N,K);
    } else if (hasb) {
        gemm_wmma<1,0,0><<<grid,blk,0,s>>>(A,lda,B,ldb,C,ldc,bias,M,N,K);
    } else {
        gemm_wmma<0,0,0><<<grid,blk,0,s>>>(A,lda,B,ldb,C,ldc,bias,M,N,K);
    }
}

// Parameter flattening assumed in setup_inputs() insertion order (recursive):
// 0:x_enc 1:x_mark_enc 2:Wv 3:bv 4:Wt 5:bt 6:gN 7:bN 8:Wp 9:bp
// layer L at 10+26*L:
//   m_f: +0 Win +1 conv_w +2 conv_b +3 Wx +4 Wdt +5 bdt +6 A_log +7 D +8 Wout
//   m_r: +9..+17 (same order)
//   +18 g1 +19 b1 +20 g2 +21 b2 +22 W1 +23 b1f +24 W2 +25 b2f
extern "C" void kernel_launch(void* const* d_in, const int* in_sizes, int n_in,
                              void* d_out, int out_size, void* d_ws, size_t ws_size,
                              hipStream_t stream)
{
    (void)in_sizes; (void)n_in; (void)out_size; (void)ws_size;
    auto F = [&](int i) -> const float* { return (const float*)d_in[i]; };

    float* ws = (float*)d_ws;
    size_t off = 0;
    float* means = ws + off; off += TOKENS;
    float* stdev = ws + off; off += TOKENS;
    float* xm    = ws + off; off += 128;
    float* ENC   = ws + off; off += (size_t)TOKENS * DM;
    float* ACC   = ws + off; off += (size_t)TOKENS * DM;
    float* T0    = ws + off; off += (size_t)TOKENS * 2 * DM;   // xz
    float* T1    = ws + off; off += (size_t)TOKENS * DM;       // uc
    float* T2    = ws + off; off += (size_t)TOKENS * DM;       // delta
    float* T3    = ws + off; off += (size_t)TOKENS * (DTRANK + 2 * DSTATE); // dbc
    float* T4    = ws + off; off += (size_t)TOKENS * DM;       // y
    float* FFN   = T0;   // aliases T0..T2 (TOKENS*2048 floats exactly)
    float* DEC   = T4;   // final projection output (TOKENS*96 <= TOKENS*512)

    const size_t encBytes = (size_t)TOKENS * DM * sizeof(float);
    const int EW = TOKENS * DM / 256;   // elementwise grid

    // ---- normalization stats + xnT (into T1) ----
    stats_kernel<<<TOKENS, 128, 0, stream>>>(F(0), means, stdev, T1);
    // ---- embedding: ENC = xnT @ Wv + bv ----
    gemm(1, 0, 0, T1, SEQ, F(2), DM, ENC, DM, F(3), TOKENS, DM, SEQ, stream);
    xm_mean_kernel<<<BATCH, 32, 0, stream>>>(F(1), xm);
    add_time_kernel<<<EW, 256, 0, stream>>>(ENC, xm, F(4), F(5));

    for (int L = 0; L < 2; ++L) {
        int base = 10 + 26 * L;
        // residual accumulator for bidirectional mamba
        (void)hipMemcpyAsync(ACC, ENC, encBytes, hipMemcpyDeviceToDevice, stream);
        for (int dir = 0; dir < 2; ++dir) {
            int mb = base + dir * 9;
            const float *Win = F(mb+0), *cw = F(mb+1), *cb = F(mb+2), *Wx = F(mb+3),
                        *Wdt = F(mb+4), *bdt = F(mb+5), *Alog = F(mb+6),
                        *Dp = F(mb+7), *Wout = F(mb+8);
            // xz = ENC @ Win
            gemm(0, 0, 0, ENC, DM, Win, 2*DM, T0, 2*DM, nullptr, TOKENS, 2*DM, DM, stream);
            // uc = silu(conv2(u))
            conv_silu_kernel<<<EW, 256, 0, stream>>>(T0, cw, cb, T1, dir);
            // dbc = uc @ Wx
            gemm(0, 0, 0, T1, DM, Wx, DTRANK + 2*DSTATE, T3, DTRANK + 2*DSTATE,
                 nullptr, TOKENS, DTRANK + 2*DSTATE, DM, stream);
            // delta = softplus(dt @ Wdt + bdt)   (dt = first 32 cols of dbc)
            gemm(1, 2, 0, T3, DTRANK + 2*DSTATE, Wdt, DM, T2, DM, bdt,
                 TOKENS, DM, DTRANK, stream);
            // selective scan + skip + SiLU gate
            scan_kernel<<<BATCH * 2, 256, 0, stream>>>(T2, T3, T1, T0, Alog, Dp, T4, dir);
            // ACC += y @ Wout
            gemm(0, 0, 1, T4, DM, Wout, DM, ACC, DM, nullptr, TOKENS, DM, DM, stream);
        }
        // x = LN(x + mamba_f + mamba_r)
        layernorm_kernel<<<TOKENS, 256, 0, stream>>>(ACC, F(base+18), F(base+19), ENC);
        // FFN
        gemm(1, 1, 0, ENC, DM, F(base+22), DFF, FFN, DFF, F(base+23), TOKENS, DFF, DM, stream);
        (void)hipMemcpyAsync(ACC, ENC, encBytes, hipMemcpyDeviceToDevice, stream);
        gemm(1, 0, 1, FFN, DFF, F(base+24), DM, ACC, DM, F(base+25), TOKENS, DM, DFF, stream);
        layernorm_kernel<<<TOKENS, 256, 0, stream>>>(ACC, F(base+20), F(base+21), ENC);
    }

    // final LN -> projection -> de-normalize/transpose
    layernorm_kernel<<<TOKENS, 256, 0, stream>>>(ENC, F(6), F(7), ACC);
    gemm(1, 0, 0, ACC, DM, F(8), PRED, DEC, PRED, F(9), TOKENS, PRED, DM, stream);
    out_kernel<<<(BATCH * PRED * NVARS + 255) / 256, 256, 0, stream>>>(
        DEC, means, stdev, (float*)d_out);
}
